// MultiHeadAttention2_70385924047429
// MI455X (gfx1250) — matrix operand order repaired and checked by reading, not compiled, not run on hardware
//
#include <hip/hip_runtime.h>

// Problem constants (from reference)
#define BB 2
#define LQN 2048
#define LKN 2048
#define EN 1024
#define HN 16
#define DHN 64

typedef _Float16 v16h __attribute__((ext_vector_type(16)));
typedef _Float16 v8h  __attribute__((ext_vector_type(8)));
typedef float    v8f  __attribute__((ext_vector_type(8)));
typedef unsigned int v4u __attribute__((ext_vector_type(4)));
typedef int      v8i  __attribute__((ext_vector_type(8)));
typedef int      v4i  __attribute__((ext_vector_type(4)));

static __device__ __forceinline__ v8f wmma_f16(v16h a, v16h b, v8f c) {
    // D = A(16x32 f16) * B(32x16 f16) + C(16x16 f32)
    return __builtin_amdgcn_wmma_f32_16x16x32_f16(
        /*neg_a=*/false, a, /*neg_b=*/false, b,
        /*c_mod=*/(short)0, c, /*reuse_a=*/false, /*reuse_b=*/false);
}

// ---------------------------------------------------------------------------
// fp32 -> f16 cast (vectorized x4)
// ---------------------------------------------------------------------------
__global__ __launch_bounds__(256)
void cast_f32_to_f16_kernel(const float* __restrict__ in,
                            _Float16* __restrict__ out, int n4) {
    int i = blockIdx.x * 256 + threadIdx.x;
    if (i < n4) {
        float4 v = reinterpret_cast<const float4*>(in)[i];
        _Float16* o = out + (size_t)i * 4;
        o[0] = (_Float16)v.x; o[1] = (_Float16)v.y;
        o[2] = (_Float16)v.z; o[3] = (_Float16)v.w;
    }
}

// ---------------------------------------------------------------------------
// LayerNorm(input_q) -> f16   (one block per row of E=1024, 256 thr x 4 elems)
// ---------------------------------------------------------------------------
__global__ __launch_bounds__(256)
void ln_q_f16_kernel(const float* __restrict__ x,
                     const float* __restrict__ gamma,
                     const float* __restrict__ beta,
                     _Float16* __restrict__ out) {
    const int row = blockIdx.x;
    const int tid = threadIdx.x;
    const float* xr = x + (size_t)row * EN;
    float4 v = reinterpret_cast<const float4*>(xr)[tid];
    float s  = v.x + v.y + v.z + v.w;
    float ss = v.x*v.x + v.y*v.y + v.z*v.z + v.w*v.w;
#pragma unroll
    for (int off = 16; off > 0; off >>= 1) {
        s  += __shfl_xor(s,  off, 32);
        ss += __shfl_xor(ss, off, 32);
    }
    __shared__ float sbuf[8], ssbuf[8];
    if ((tid & 31) == 0) { sbuf[tid >> 5] = s; ssbuf[tid >> 5] = ss; }
    __syncthreads();
    float S = 0.f, SS = 0.f;
#pragma unroll
    for (int i = 0; i < 8; ++i) { S += sbuf[i]; SS += ssbuf[i]; }
    const float mean = S * (1.0f / EN);
    const float var  = SS * (1.0f / EN) - mean * mean;
    const float rstd = rsqrtf(var + 1e-6f);
    float4 g = reinterpret_cast<const float4*>(gamma)[tid];
    float4 bt = reinterpret_cast<const float4*>(beta)[tid];
    _Float16* o = out + (size_t)row * EN + tid * 4;
    o[0] = (_Float16)((v.x - mean) * rstd * g.x + bt.x);
    o[1] = (_Float16)((v.y - mean) * rstd * g.y + bt.y);
    o[2] = (_Float16)((v.z - mean) * rstd * g.z + bt.z);
    o[3] = (_Float16)((v.w - mean) * rstd * g.w + bt.w);
}

// ---------------------------------------------------------------------------
// Post-LayerNorm, in place on fp32 output rows
// ---------------------------------------------------------------------------
__global__ __launch_bounds__(256)
void ln_post_kernel(float* __restrict__ y,
                    const float* __restrict__ gamma,
                    const float* __restrict__ beta) {
    const int row = blockIdx.x;
    const int tid = threadIdx.x;
    float* yr = y + (size_t)row * EN;
    float4 v = reinterpret_cast<const float4*>(yr)[tid];
    float s  = v.x + v.y + v.z + v.w;
    float ss = v.x*v.x + v.y*v.y + v.z*v.z + v.w*v.w;
#pragma unroll
    for (int off = 16; off > 0; off >>= 1) {
        s  += __shfl_xor(s,  off, 32);
        ss += __shfl_xor(ss, off, 32);
    }
    __shared__ float sbuf[8], ssbuf[8];
    if ((tid & 31) == 0) { sbuf[tid >> 5] = s; ssbuf[tid >> 5] = ss; }
    __syncthreads();
    float S = 0.f, SS = 0.f;
#pragma unroll
    for (int i = 0; i < 8; ++i) { S += sbuf[i]; SS += ssbuf[i]; }
    const float mean = S * (1.0f / EN);
    const float var  = SS * (1.0f / EN) - mean * mean;
    const float rstd = rsqrtf(var + 1e-6f);
    float4 g = reinterpret_cast<const float4*>(gamma)[tid];
    float4 bt = reinterpret_cast<const float4*>(beta)[tid];
    float4 o;
    o.x = (v.x - mean) * rstd * g.x + bt.x;
    o.y = (v.y - mean) * rstd * g.y + bt.y;
    o.z = (v.z - mean) * rstd * g.z + bt.z;
    o.w = (v.w - mean) * rstd * g.w + bt.w;
    reinterpret_cast<float4*>(yr)[tid] = o;
}

// ---------------------------------------------------------------------------
// WMMA GEMM: C[M,N] = A[M,K]f16 @ W[K,N]f16 + bias, fused epilogues.
// Block = 128 threads (4 waves), tile = 64x64, BK = 32.
//   MODE 0: q-proj  -> scale 1/8, write f16 into qh [B,H,LQ,DH]
//   MODE 1: kv-proj -> split K/V,  write f16 into kh/vh [B,H,LK,DH]
//   MODE 2: out-proj -> + bias + residual(input_q), write fp32 (pre-LN) to out
// ---------------------------------------------------------------------------
template <int MODE>
__global__ __launch_bounds__(128)
void gemm_wmma_kernel(const _Float16* __restrict__ A,
                      const _Float16* __restrict__ W,
                      const float* __restrict__ bias,
                      _Float16* __restrict__ out_a,
                      _Float16* __restrict__ out_b,
                      const float* __restrict__ resid,
                      float* __restrict__ out_f32,
                      int M, int N, int K) {
    __shared__ alignas(32) _Float16 As[64][32];     // [m][k]
    __shared__ alignas(32) _Float16 BsT[64][32];    // [n][k]  (K-transposed)

    const int tid  = threadIdx.x;
    const int wave = tid >> 5;
    const int lane = tid & 31;
    const int hf   = lane >> 4;   // lane half (0/1)
    const int l16  = lane & 15;
    const int m0   = blockIdx.y * 64;
    const int n0   = blockIdx.x * 64;

    const v8f zero8 = {0.f,0.f,0.f,0.f,0.f,0.f,0.f,0.f};
    v8f acc[4] = {zero8, zero8, zero8, zero8};

    for (int k0 = 0; k0 < K; k0 += 32) {
        __syncthreads();
        {   // A tile 64x32, each thread 16 contiguous halfs
            const int row = tid >> 1;
            const int seg = (tid & 1) * 16;
            const _Float16* src = A + (size_t)(m0 + row) * K + k0 + seg;
            v8h a0 = *reinterpret_cast<const v8h*>(src);
            v8h a1 = *reinterpret_cast<const v8h*>(src + 8);
            *reinterpret_cast<v8h*>(&As[row][seg])     = a0;
            *reinterpret_cast<v8h*>(&As[row][seg + 8]) = a1;
        }
        {   // W tile 32x64, stored transposed -> BsT[n][k]
            const int kl  = tid >> 2;
            const int seg = (tid & 3) * 16;
            const _Float16* src = W + (size_t)(k0 + kl) * N + n0 + seg;
            v8h b0 = *reinterpret_cast<const v8h*>(src);
            v8h b1 = *reinterpret_cast<const v8h*>(src + 8);
#pragma unroll
            for (int i = 0; i < 8; ++i) BsT[seg + i][kl] = b0[i];
#pragma unroll
            for (int i = 0; i < 8; ++i) BsT[seg + 8 + i][kl] = b1[i];
        }
        __syncthreads();

        // A fragment for this wave's 16 rows (contiguous k = 16*hf + s, matching B)
        v16h afrag;
        {
            const int m = wave * 16 + l16;
            v8h lo = *reinterpret_cast<const v8h*>(&As[m][hf * 16]);
            v8h hi = *reinterpret_cast<const v8h*>(&As[m][hf * 16 + 8]);
#pragma unroll
            for (int i = 0; i < 8; ++i) { afrag[i] = lo[i]; afrag[8 + i] = hi[i]; }
        }
#pragma unroll
        for (int nt = 0; nt < 4; ++nt) {
            const int n = nt * 16 + l16;
            v16h bfrag = *reinterpret_cast<const v16h*>(&BsT[n][hf * 16]);
            acc[nt] = wmma_f16(afrag, bfrag, acc[nt]);
        }
    }

    // Epilogue (C layout: VGPR r -> M = r + 8*hf; col = lane&15 within tile)
#pragma unroll
    for (int nt = 0; nt < 4; ++nt) {
        const int col = n0 + nt * 16 + l16;
        const float bv = bias[col];
#pragma unroll
        for (int r = 0; r < 8; ++r) {
            const int row = m0 + wave * 16 + hf * 8 + r;
            float v = acc[nt][r] + bv;
            if (MODE == 0) {
                v *= 0.125f;  // fold 1/sqrt(DH)
                const int b  = row >> 11, lq = row & 2047;
                const int h  = col >> 6,  dh = col & 63;
                out_a[(((size_t)(b * HN + h)) * LQN + lq) * DHN + dh] = (_Float16)v;
            } else if (MODE == 1) {
                const int b = row >> 11, lk = row & 2047;
                if (col < EN) {
                    const int h = col >> 6, dh = col & 63;
                    out_a[(((size_t)(b * HN + h)) * LKN + lk) * DHN + dh] = (_Float16)v;
                } else {
                    const int c2 = col - EN;
                    const int h = c2 >> 6, dh = c2 & 63;
                    out_b[(((size_t)(b * HN + h)) * LKN + lk) * DHN + dh] = (_Float16)v;
                }
            } else {
                v += resid[(size_t)row * EN + col];
                out_f32[(size_t)row * EN + col] = v;
            }
        }
    }
}

// ---------------------------------------------------------------------------
// Fused flash attention: block = 4 waves, each wave owns a 16-row Q tile of a
// shared (b,h, 64-row) block; K/V streamed in 32-key chunks through LDS.
// K tile is fetched by the Tensor Data Mover (TENSOR_LOAD_TO_LDS, wave 0),
// V tile is staged transposed by VALU so the ctx-WMMA B fragment is one
// contiguous 32B ds_load. Q pre-scaled by 1/8; mask!=0 -> -1e30.
// ---------------------------------------------------------------------------
__global__ __launch_bounds__(128)
void attn_flash_kernel(const _Float16* __restrict__ qh,
                       const _Float16* __restrict__ kh,
                       const _Float16* __restrict__ vh,
                       const int* __restrict__ mask,
                       _Float16* __restrict__ ctx) {
    __shared__ alignas(32) _Float16 Ks[32][64];       // [key_local][dh] (TDM dest)
    __shared__ alignas(32) _Float16 VsT[64][32];      // [dh][key_local]
    __shared__ alignas(32) _Float16 Pl[4][16][32];    // per-wave P scratch

    const int tid  = threadIdx.x;
    const int wave = tid >> 5;
    const int lane = tid & 31;
    const int hf   = lane >> 4;
    const int l16  = lane & 15;
    const int bh   = blockIdx.x >> 5;           // 32 q-blocks per (b,h)
    const int qt   = blockIdx.x & 31;
    const int b    = bh >> 4;
    const int h    = bh & 15;
    const int q0   = qt * 64 + wave * 16;

    // --- TDM descriptor group1 (loop invariant): 2-byte elems, 2D tensor,
    //     tensor_dim0=64 (DH), tensor_dim1=LK, tile 64x32, row stride 64.
    v8i g1k;
    g1k[0] = (1 << 16);            // data_size = 2 bytes, no barrier/iter/pad
    g1k[1] = (int)(64u << 16);     // tensor_dim0[15:0] = 64
    g1k[2] = (int)(2048u << 16);   // tensor_dim1[15:0] = 2048
    g1k[3] = (int)(64u << 16);     // tile_dim0 = 64
    g1k[4] = 32;                   // tile_dim1 = 32, tile_dim2 = 0
    g1k[5] = 64;                   // tensor_dim0_stride = 64 elements
    g1k[6] = 0;
    g1k[7] = 0;
    const v4i z4 = {0, 0, 0, 0};
    const v8i z8 = {0, 0, 0, 0, 0, 0, 0, 0};
    const unsigned ldsK = (unsigned)(uintptr_t)(&Ks[0][0]);

    // Q fragments (dh 0..31 and 32..63), loaded once
    v16h qf[2];
    {
        const _Float16* qrow = qh + ((size_t)bh * LQN + q0 + l16) * DHN;
#pragma unroll
        for (int kc = 0; kc < 2; ++kc) {
            v8h lo = *reinterpret_cast<const v8h*>(qrow + kc * 32 + hf * 16);
            v8h hi = *reinterpret_cast<const v8h*>(qrow + kc * 32 + hf * 16 + 8);
#pragma unroll
            for (int i = 0; i < 8; ++i) { qf[kc][i] = lo[i]; qf[kc][8 + i] = hi[i]; }
        }
    }

    const v8f zero8 = {0.f,0.f,0.f,0.f,0.f,0.f,0.f,0.f};
    v8f acc[4] = {zero8, zero8, zero8, zero8};   // 16 x 64 context accumulator
    float mrun[8], lrun[8];
#pragma unroll
    for (int r = 0; r < 8; ++r) { mrun[r] = -1e30f; lrun[r] = 0.f; }

    for (int j0 = 0; j0 < LKN; j0 += 32) {
        __syncthreads();   // previous tile fully consumed before overwrite
        if (wave == 0) {
            // TDM: DMA the 32x64 f16 K chunk into LDS (issued once per block)
            const size_t gaddr = (size_t)(const void*)(kh + ((size_t)bh * LKN + j0) * DHN);
            v4u g0;
            g0[0] = 1u;                                   // count=1, user mode
            g0[1] = ldsK;                                 // lds_addr
            g0[2] = (unsigned)(gaddr & 0xFFFFFFFFull);    // global_addr[31:0]
            g0[3] = (unsigned)((gaddr >> 32) & 0x1FFFFFFu) | (2u << 30); // type=2
            __builtin_amdgcn_tensor_load_to_lds(g0, g1k, z4, z4, z8, 0);
        }
        {   // cooperative V chunk load, stored transposed
            const int kl  = tid >> 2;
            const int seg = (tid & 3) * 16;
            const _Float16* vs = vh + ((size_t)bh * LKN + j0 + kl) * DHN + seg;
            v8h v0v = *reinterpret_cast<const v8h*>(vs);
            v8h v1v = *reinterpret_cast<const v8h*>(vs + 8);
#pragma unroll
            for (int i = 0; i < 8; ++i) VsT[seg + i][kl] = v0v[i];
#pragma unroll
            for (int i = 0; i < 8; ++i) VsT[seg + 8 + i][kl] = v1v[i];
        }
        if (wave == 0) __builtin_amdgcn_s_wait_tensorcnt(0);
        __syncthreads();   // K (TDM) + V (VALU) tiles visible to all waves

        // Scores: two 16x16 tiles (keys j0.. and j0+16..), K accumulated over DH=64
        float s[2][8];
#pragma unroll
        for (int t = 0; t < 2; ++t) {
            v16h kb0 = *reinterpret_cast<const v16h*>(&Ks[t * 16 + l16][hf * 16]);
            v16h kb1 = *reinterpret_cast<const v16h*>(&Ks[t * 16 + l16][32 + hf * 16]);
            v8f sv = zero8;
            sv = wmma_f16(qf[0], kb0, sv);
            sv = wmma_f16(qf[1], kb1, sv);
            const int keycol = j0 + t * 16 + l16;
#pragma unroll
            for (int r = 0; r < 8; ++r) {
                const int qrow = q0 + hf * 8 + r;
                const int mv = mask[((size_t)b * LQN + qrow) * LKN + keycol];
                s[t][r] = mv ? -1e30f : sv[r];
            }
        }

        // Online softmax over this 32-key chunk (row = 16-lane group)
#pragma unroll
        for (int r = 0; r < 8; ++r) {
            float mx = fmaxf(s[0][r], s[1][r]);
#pragma unroll
            for (int off = 8; off > 0; off >>= 1) mx = fmaxf(mx, __shfl_xor(mx, off));
            const float mnew = fmaxf(mrun[r], mx);
            const float corr = __expf(mrun[r] - mnew);
            mrun[r] = mnew;
            const float p0 = __expf(s[0][r] - mnew);
            const float p1 = __expf(s[1][r] - mnew);
            float rs = p0 + p1;
#pragma unroll
            for (int off = 8; off > 0; off >>= 1) rs += __shfl_xor(rs, off);
            lrun[r] = lrun[r] * corr + rs;
#pragma unroll
            for (int dt = 0; dt < 4; ++dt) acc[dt][r] *= corr;
            const int prow = hf * 8 + r;
            Pl[wave][prow][l16]      = (_Float16)p0;
            Pl[wave][prow][16 + l16] = (_Float16)p1;
        }

        // ctx += P(16x32) @ V(32x64)
        v16h pf;
        {
            v8h lo = *reinterpret_cast<const v8h*>(&Pl[wave][l16][hf * 16]);
            v8h hi = *reinterpret_cast<const v8h*>(&Pl[wave][l16][hf * 16 + 8]);
#pragma unroll
            for (int i = 0; i < 8; ++i) { pf[i] = lo[i]; pf[8 + i] = hi[i]; }
        }
#pragma unroll
        for (int dt = 0; dt < 4; ++dt) {
            v16h vb = *reinterpret_cast<const v16h*>(&VsT[dt * 16 + l16][hf * 16]);
            acc[dt] = wmma_f16(pf, vb, acc[dt]);
        }
    }

    // Normalize and write ctx f16 in [B,LQ,E] layout
#pragma unroll
    for (int dt = 0; dt < 4; ++dt) {
#pragma unroll
        for (int r = 0; r < 8; ++r) {
            const int qrow = q0 + hf * 8 + r;
            const float v = acc[dt][r] / lrun[r];
            ctx[((size_t)b * LQN + qrow) * EN + h * DHN + dt * 16 + l16] = (_Float16)v;
        }
    }
}

// ---------------------------------------------------------------------------
// Host-side orchestration
// ---------------------------------------------------------------------------
extern "C" void kernel_launch(void* const* d_in, const int* in_sizes, int n_in,
                              void* d_out, int out_size, void* d_ws, size_t ws_size,
                              hipStream_t stream) {
    (void)in_sizes; (void)n_in; (void)out_size; (void)ws_size;
    const float* input_q    = (const float*)d_in[0];
    const float* input_kv   = (const float*)d_in[1];
    const float* Wq         = (const float*)d_in[2];
    const float* bq         = (const float*)d_in[3];
    const float* Wkv        = (const float*)d_in[4];
    const float* bkv        = (const float*)d_in[5];
    const float* Wout       = (const float*)d_in[6];
    const float* bout       = (const float*)d_in[7];
    const float* gamma_q    = (const float*)d_in[8];
    const float* beta_q     = (const float*)d_in[9];
    const float* gamma_post = (const float*)d_in[10];
    const float* beta_post  = (const float*)d_in[11];
    const int*   mask       = (const int*)d_in[12];
    float* out = (float*)d_out;

    // Workspace carve-out (56 MiB total)
    char* ws = (char*)d_ws;
    _Float16* nq    = (_Float16*)(ws);                       //  8 MiB [B*LQ,E]
    _Float16* kvh   = (_Float16*)(ws + ((size_t)8  << 20));  //  8 MiB [B*LK,E]
    _Float16* WqH   = (_Float16*)(ws + ((size_t)16 << 20));  //  2 MiB
    _Float16* WkvH  = (_Float16*)(ws + ((size_t)18 << 20));  //  4 MiB
    _Float16* WoutH = (_Float16*)(ws + ((size_t)22 << 20));  //  2 MiB
    _Float16* qh    = (_Float16*)(ws + ((size_t)24 << 20));  //  8 MiB [B,H,LQ,DH]
    _Float16* kh    = (_Float16*)(ws + ((size_t)32 << 20));  //  8 MiB
    _Float16* vh    = (_Float16*)(ws + ((size_t)40 << 20));  //  8 MiB
    _Float16* ctxh  = (_Float16*)(ws + ((size_t)48 << 20));  //  8 MiB [B*LQ,E]

    // 1) f32 -> f16 casts (weights + input_kv)
    cast_f32_to_f16_kernel<<<(EN * EN) / 4 / 256, 256, 0, stream>>>(Wq, WqH, (EN * EN) / 4);
    cast_f32_to_f16_kernel<<<(EN * 2 * EN) / 4 / 256, 256, 0, stream>>>(Wkv, WkvH, (EN * 2 * EN) / 4);
    cast_f32_to_f16_kernel<<<(EN * EN) / 4 / 256, 256, 0, stream>>>(Wout, WoutH, (EN * EN) / 4);
    cast_f32_to_f16_kernel<<<(BB * LKN * EN) / 4 / 256, 256, 0, stream>>>(input_kv, kvh, (BB * LKN * EN) / 4);

    // 2) pre-norm on Q path -> f16
    ln_q_f16_kernel<<<BB * LQN, 256, 0, stream>>>(input_q, gamma_q, beta_q, nq);

    // 3) projections (WMMA GEMMs, fused scale / head-layout epilogues)
    gemm_wmma_kernel<0><<<dim3(EN / 64, BB * LQN / 64), 128, 0, stream>>>(
        nq, WqH, bq, qh, nullptr, nullptr, nullptr, BB * LQN, EN, EN);
    gemm_wmma_kernel<1><<<dim3(2 * EN / 64, BB * LKN / 64), 128, 0, stream>>>(
        kvh, WkvH, bkv, kh, vh, nullptr, nullptr, BB * LKN, 2 * EN, EN);

    // 4) fused flash attention (WMMA + TDM K-tile streaming)
    attn_flash_kernel<<<BB * HN * (LQN / 64), 128, 0, stream>>>(qh, kh, vh, mask, ctxh);

    // 5) output projection + bias + residual -> d_out (pre-LN fp32)
    gemm_wmma_kernel<2><<<dim3(EN / 64, BB * LQN / 64), 128, 0, stream>>>(
        ctxh, WoutH, bout, nullptr, nullptr, input_q, out, BB * LQN, EN, EN);

    // 6) post layer-norm, in place on d_out
    ln_post_kernel<<<BB * LQN, 256, 0, stream>>>(out, gamma_post, beta_post);
}
